// Attention_28716151341673
// MI455X (gfx1250) — matrix operand order repaired and checked by reading, not compiled, not run on hardware
//
#include <hip/hip_runtime.h>
#include <hip/hip_bf16.h>

typedef _Float16 h8  __attribute__((ext_vector_type(8)));
typedef _Float16 v16h __attribute__((ext_vector_type(16)));
typedef float    v8f  __attribute__((ext_vector_type(8)));
typedef int      i4   __attribute__((ext_vector_type(4)));

#define HID   1024
#define NH    16
#define HD    64
#define BATCH 8
#define LW    512
#define LE    64
#define SEQ   576          // LW + LE
#define MROWS (BATCH*SEQ)  // 4608

#if __has_builtin(__builtin_amdgcn_global_load_async_to_lds_b128) && \
    __has_builtin(__builtin_amdgcn_s_wait_asynccnt)
#define HAVE_ASYNC_LDS 1
#else
#define HAVE_ASYNC_LDS 0
#endif

// 16-byte global -> LDS copy (async on CDNA5, fallback = through-register).
static __device__ __forceinline__ void cp16_g2l(const _Float16* g, _Float16* l) {
#if HAVE_ASYNC_LDS
    __builtin_amdgcn_global_load_async_to_lds_b128(
        (__attribute__((address_space(1))) i4*)(g),
        (__attribute__((address_space(3))) i4*)(l), 0, 0);
#else
    *(h8*)l = *(const h8*)g;
#endif
}

#if HAVE_ASYNC_LDS
#define ASYNC_WAIT(N) __builtin_amdgcn_s_wait_asynccnt(N)
#else
#define ASYNC_WAIT(N) __syncthreads()
#endif

// Assemble a 16-half A/B fragment from two 16-byte loads.
static __device__ __forceinline__ v16h ld_frag(const _Float16* p0, const _Float16* p1) {
    union { v16h v; h8 h[2]; } u;
    u.h[0] = *(const h8*)p0;
    u.h[1] = *(const h8*)p1;
    return u.v;
}

static __device__ __forceinline__ v8f wmma_f16(v16h a, v16h b, v8f c) {
    return __builtin_amdgcn_wmma_f32_16x16x32_f16(false, a, false, b, (short)0, c, false, false);
}

// ---------------- prep: concat(word, entity) -> f16 row-major [M][H] --------
__global__ void __launch_bounds__(256)
k_convert_x(const float* __restrict__ word,
            const float* __restrict__ ent,
            _Float16* __restrict__ xh) {
    unsigned idx = blockIdx.x * 256u + threadIdx.x;      // < MROWS*HID
    unsigned b = idx / (SEQ * HID);
    unsigned r = idx % (SEQ * HID);
    unsigned s = r >> 10;
    unsigned d = r & 1023;
    float v = (s < LW) ? word[((size_t)(b * LW + s)) * HID + d]
                       : ent [((size_t)(b * LE + (s - LW))) * HID + d];
    xh[idx] = (_Float16)v;
}

// ---------------- prep: Wt[n][k] = (f16) W[k][n] ----------------------------
__global__ void __launch_bounds__(256)
k_transpose_w(const float* __restrict__ W, _Float16* __restrict__ Wt) {
    unsigned idx = blockIdx.x * 256u + threadIdx.x;      // < HID*HID
    unsigned n = idx >> 10;
    unsigned k = idx & 1023;
    Wt[idx] = (_Float16)W[(size_t)k * HID + n];
}

// ---------------- QKV projection GEMM: one wave = 64 rows x 64 cols ---------
// A-tile (64x32 f16) double-buffered in LDS via async global->LDS copies.
// vmode==0: out[((b*NH+h)*SEQ + s)*HD + d]      (q, k)
// vmode==1: out[((b*NH+h)*HD + d)*SEQ + s]      (v transposed)
__global__ void __launch_bounds__(32, 1)
k_gemm_qkv(const _Float16* __restrict__ A,
           const _Float16* __restrict__ Wt,
           const float* __restrict__ bias,
           _Float16* __restrict__ out, int vmode) {
    __shared__ _Float16 atile[2][64 * 32];

    const int lane  = threadIdx.x;
    const int tileM = blockIdx.x;          // 0..71
    const int head  = blockIdx.y;          // 0..15
    const int r0 = tileM * 64;
    const int b  = r0 / SEQ;               // SEQ = 9*64, tiles never cross batch
    const int s0 = r0 % SEQ;

    const int rowL = lane >> 2;            // async-copy lane mapping
    const int colH = (lane & 3) * 8;

    const int aRow = lane & 15;
    const int aOff = (lane & 16) ? 8 : 0;
    const int bCol = lane & 15;
    const int bOff = (lane & 16) ? 8 : 0;

    v8f c[4][4];
#pragma unroll
    for (int mi = 0; mi < 4; ++mi)
#pragma unroll
        for (int t = 0; t < 4; ++t) c[mi][t] = v8f{};

    // prologue: stage first A tile
#pragma unroll
    for (int j = 0; j < 8; ++j)
        cp16_g2l(A + (size_t)(r0 + j * 8 + rowL) * HID + colH,
                 &atile[0][(j * 8 + rowL) * 32 + colH]);

    for (int k0 = 0; k0 < HID; k0 += 32) {
        const int cur = (k0 >> 5) & 1;
        if (k0 + 32 < HID) {
#pragma unroll
            for (int j = 0; j < 8; ++j)
                cp16_g2l(A + (size_t)(r0 + j * 8 + rowL) * HID + (k0 + 32) + colH,
                         &atile[cur ^ 1][(j * 8 + rowL) * 32 + colH]);
            ASYNC_WAIT(8);     // retire current tile (in-order completion)
        } else {
            ASYNC_WAIT(0);
        }

        v16h af[4];
#pragma unroll
        for (int mi = 0; mi < 4; ++mi) {
            const _Float16* ap = &atile[cur][(mi * 16 + aRow) * 32 + aOff];
            af[mi] = ld_frag(ap, ap + 16);
        }
#pragma unroll
        for (int t = 0; t < 4; ++t) {
            const _Float16* bp = Wt + (size_t)(head * 64 + t * 16 + bCol) * HID + k0 + bOff;
            v16h bf = ld_frag(bp, bp + 16);
#pragma unroll
            for (int mi = 0; mi < 4; ++mi)
                c[mi][t] = wmma_f16(af[mi], bf, c[mi][t]);
        }
    }

    const int col   = lane & 15;
    const int mbase = (lane & 16) ? 8 : 0;
#pragma unroll
    for (int mi = 0; mi < 4; ++mi) {
#pragma unroll
        for (int t = 0; t < 4; ++t) {
            const int nG = head * 64 + t * 16 + col;
            const float bv = bias[nG];
            if (!vmode) {
#pragma unroll
                for (int i = 0; i < 8; ++i) {
                    int s = s0 + mi * 16 + mbase + i;
                    out[((size_t)(b * NH + head) * SEQ + s) * HD + (t * 16 + col)] =
                        (_Float16)(c[mi][t][i] + bv);
                }
            } else {
                h8 hv;
#pragma unroll
                for (int i = 0; i < 8; ++i) hv[i] = (_Float16)(c[mi][t][i] + bv);
                *(h8*)&out[((size_t)(b * NH + head) * HD + (t * 16 + col)) * SEQ +
                           s0 + mi * 16 + mbase] = hv;
            }
        }
    }
}

// ---------------- attention: scores -> softmax -> ctx -----------------------
__global__ void __launch_bounds__(32, 1)
k_attn(const _Float16* __restrict__ qh,
       const _Float16* __restrict__ kh,
       const _Float16* __restrict__ vTh,
       const float* __restrict__ mask,
       _Float16* __restrict__ ctxh) {
    __shared__ float    sc[16 * SEQ];      // 36 KB
    __shared__ _Float16 pr[16 * SEQ];      // 18 KB

    const int lane = threadIdx.x;
    const int qt   = blockIdx.x;           // 0..35
    const int head = blockIdx.y;           // 0..15
    const int b    = blockIdx.z;           // 0..7
    const int bh   = b * NH + head;

    const int aRow = lane & 15;
    const int aOff = (lane & 16) ? 8 : 0;
    const int bCol = lane & 15;
    const int bOff = (lane & 16) ? 8 : 0;
    const int col  = lane & 15;
    const int mb   = (lane & 16) ? 8 : 0;

    // Preload the 16x64 q tile as two A fragments (K=0..31, K=32..63).
    const _Float16* qrow = qh + ((size_t)(bh * SEQ + qt * 16 + aRow)) * HD;
    v16h a0 = ld_frag(qrow + aOff,      qrow + aOff + 16);
    v16h a1 = ld_frag(qrow + 32 + aOff, qrow + 32 + aOff + 16);

    // Phase 1: scores = q @ k^T / 8 + mask  -> LDS (f32)
    for (int kt = 0; kt < 36; ++kt) {
        v8f c = v8f{};
        const _Float16* kp0 = kh + ((size_t)(bh * SEQ + kt * 16 + bCol)) * HD + bOff;
        c = wmma_f16(a0, ld_frag(kp0,      kp0 + 16),      c);
        c = wmma_f16(a1, ld_frag(kp0 + 32, kp0 + 32 + 16), c);
        const int key = kt * 16 + col;
        const float mval = mask[b * SEQ + key];
#pragma unroll
        for (int i = 0; i < 8; ++i)
            sc[(mb + i) * SEQ + key] = c[i] * 0.125f + mval;
    }
    __syncthreads();

    // Phase 2: softmax, write probs as f16 row-major in LDS.
    if (lane < 16) {
        const int row = lane;
        float mx = -3.0e38f;
        for (int j = 0; j < SEQ; ++j) mx = fmaxf(mx, sc[row * SEQ + j]);
        float sum = 0.0f;
        for (int j = 0; j < SEQ; ++j) {
            float e = __expf(sc[row * SEQ + j] - mx);
            sc[row * SEQ + j] = e;
            sum += e;
        }
        float inv = 1.0f / sum;
        for (int j = 0; j < SEQ; ++j)
            pr[row * SEQ + j] = (_Float16)(sc[row * SEQ + j] * inv);
    }
    __syncthreads();

    // Phase 3: ctx = probs @ v   (v is stored transposed: [bh][d][s])
    v8f o[4] = {v8f{}, v8f{}, v8f{}, v8f{}};
    for (int k0 = 0; k0 < SEQ; k0 += 32) {
        v16h af = ld_frag(&pr[aRow * SEQ + k0 + aOff], &pr[aRow * SEQ + k0 + aOff + 16]);
#pragma unroll
        for (int t = 0; t < 4; ++t) {
            const _Float16* vp = vTh + ((size_t)bh * HD + t * 16 + bCol) * SEQ + k0 + bOff;
            o[t] = wmma_f16(af, ld_frag(vp, vp + 16), o[t]);
        }
    }
#pragma unroll
    for (int t = 0; t < 4; ++t)
#pragma unroll
        for (int i = 0; i < 8; ++i)
            ctxh[((size_t)(b * SEQ + qt * 16 + mb + i)) * HID + head * 64 + t * 16 + col] =
                (_Float16)o[t][i];
}

// ---------------- output projection + bias + residual (64x64 tiles) ---------
__global__ void __launch_bounds__(32, 1)
k_outproj(const _Float16* __restrict__ ctxh,
          const _Float16* __restrict__ Wto,
          const float* __restrict__ bo,
          const float* __restrict__ word,
          const float* __restrict__ ent,
          float* __restrict__ y) {
    __shared__ _Float16 atile[2][64 * 32];

    const int lane  = threadIdx.x;
    const int tileM = blockIdx.x;          // 0..71
    const int head  = blockIdx.y;          // 64-col strip index
    const int r0 = tileM * 64;
    const int b  = r0 / SEQ;
    const int s0 = r0 % SEQ;

    const int rowL = lane >> 2;
    const int colH = (lane & 3) * 8;

    const int aRow = lane & 15;
    const int aOff = (lane & 16) ? 8 : 0;
    const int bCol = lane & 15;
    const int bOff = (lane & 16) ? 8 : 0;

    v8f c[4][4];
#pragma unroll
    for (int mi = 0; mi < 4; ++mi)
#pragma unroll
        for (int t = 0; t < 4; ++t) c[mi][t] = v8f{};

#pragma unroll
    for (int j = 0; j < 8; ++j)
        cp16_g2l(ctxh + (size_t)(r0 + j * 8 + rowL) * HID + colH,
                 &atile[0][(j * 8 + rowL) * 32 + colH]);

    for (int k0 = 0; k0 < HID; k0 += 32) {
        const int cur = (k0 >> 5) & 1;
        if (k0 + 32 < HID) {
#pragma unroll
            for (int j = 0; j < 8; ++j)
                cp16_g2l(ctxh + (size_t)(r0 + j * 8 + rowL) * HID + (k0 + 32) + colH,
                         &atile[cur ^ 1][(j * 8 + rowL) * 32 + colH]);
            ASYNC_WAIT(8);
        } else {
            ASYNC_WAIT(0);
        }

        v16h af[4];
#pragma unroll
        for (int mi = 0; mi < 4; ++mi) {
            const _Float16* ap = &atile[cur][(mi * 16 + aRow) * 32 + aOff];
            af[mi] = ld_frag(ap, ap + 16);
        }
#pragma unroll
        for (int t = 0; t < 4; ++t) {
            const _Float16* bp = Wto + (size_t)(head * 64 + t * 16 + bCol) * HID + k0 + bOff;
            v16h bf = ld_frag(bp, bp + 16);
#pragma unroll
            for (int mi = 0; mi < 4; ++mi)
                c[mi][t] = wmma_f16(af[mi], bf, c[mi][t]);
        }
    }

    const int col   = lane & 15;
    const int mbase = (lane & 16) ? 8 : 0;
#pragma unroll
    for (int mi = 0; mi < 4; ++mi) {
#pragma unroll
        for (int t = 0; t < 4; ++t) {
            const int nG = head * 64 + t * 16 + col;
            const float bv = bo[nG];
#pragma unroll
            for (int i = 0; i < 8; ++i) {
                int s = s0 + mi * 16 + mbase + i;
                float res = (s < LW) ? word[((size_t)(b * LW + s)) * HID + nG]
                                     : ent [((size_t)(b * LE + (s - LW))) * HID + nG];
                y[((size_t)(r0 + mi * 16 + mbase + i)) * HID + nG] = c[mi][t][i] + bv + res;
            }
        }
    }
}

// ---------------- per-token LayerNorm + word/entity split -------------------
__global__ void __launch_bounds__(256)
k_layernorm(const float* __restrict__ y,
            const float* __restrict__ lnw,
            const float* __restrict__ lnb,
            float* __restrict__ outp) {
    __shared__ float red[256];
    const int tid = threadIdx.x;
    const int tok = blockIdx.x;            // 0..MROWS-1
    const float* base = y + (size_t)tok * HID;

    float xv[4];
    float lsum = 0.0f;
#pragma unroll
    for (int j = 0; j < 4; ++j) { xv[j] = base[tid + j * 256]; lsum += xv[j]; }
    red[tid] = lsum; __syncthreads();
    for (int off = 128; off > 0; off >>= 1) {
        if (tid < off) red[tid] += red[tid + off];
        __syncthreads();
    }
    const float mean = red[0] * (1.0f / HID);
    __syncthreads();

    float lvar = 0.0f;
#pragma unroll
    for (int j = 0; j < 4; ++j) { float d = xv[j] - mean; lvar += d * d; }
    red[tid] = lvar; __syncthreads();
    for (int off = 128; off > 0; off >>= 1) {
        if (tid < off) red[tid] += red[tid + off];
        __syncthreads();
    }
    const float inv = rsqrtf(red[0] * (1.0f / HID) + 1e-12f);

    const int b = tok / SEQ;
    const int s = tok % SEQ;
    float* dst = (s < LW)
        ? outp + ((size_t)(b * LW + s)) * HID
        : outp + (size_t)BATCH * LW * HID + ((size_t)(b * LE + (s - LW))) * HID;
#pragma unroll
    for (int j = 0; j < 4; ++j) {
        int d = tid + j * 256;
        dst[d] = lnw[d] * ((xv[j] - mean) * inv) + lnb[d];
    }
}

extern "C" void kernel_launch(void* const* d_in, const int* in_sizes, int n_in,
                              void* d_out, int out_size, void* d_ws, size_t ws_size,
                              hipStream_t stream) {
    const float* word = (const float*)d_in[0];
    const float* ent  = (const float*)d_in[1];
    const float* mask = (const float*)d_in[2];
    const float* Wq = (const float*)d_in[3];  const float* bq = (const float*)d_in[4];
    const float* Wk = (const float*)d_in[5];  const float* bk = (const float*)d_in[6];
    const float* Wv = (const float*)d_in[7];  const float* bv = (const float*)d_in[8];
    const float* Wo = (const float*)d_in[9];  const float* bo = (const float*)d_in[10];
    const float* lnw = (const float*)d_in[11];
    const float* lnb = (const float*)d_in[12];
    float* outp = (float*)d_out;

    // Workspace carve-up (256B aligned).
    char* ws = (char*)d_ws;
    size_t off = 0;
    auto alloc = [&](size_t bytes) -> void* {
        void* p = ws + off;
        off += (bytes + 255) & ~(size_t)255;
        return p;
    };
    const size_t XE = (size_t)MROWS * HID;               // 4,718,592
    _Float16* xh   = (_Float16*)alloc(XE * 2);
    _Float16* wtq  = (_Float16*)alloc((size_t)HID * HID * 2);
    _Float16* wtk  = (_Float16*)alloc((size_t)HID * HID * 2);
    _Float16* wtv  = (_Float16*)alloc((size_t)HID * HID * 2);
    _Float16* wto  = (_Float16*)alloc((size_t)HID * HID * 2);
    _Float16* qh   = (_Float16*)alloc(XE * 2);
    _Float16* kh   = (_Float16*)alloc(XE * 2);
    _Float16* vTh  = (_Float16*)alloc(XE * 2);
    _Float16* ctxh = (_Float16*)alloc(XE * 2);
    float*    yws  = (float*)   alloc(XE * 4);
    (void)ws_size; (void)in_sizes; (void)n_in; (void)out_size;

    // 1) convert/concat x, transpose+convert weights to f16
    k_convert_x<<<dim3(XE / 256), dim3(256), 0, stream>>>(word, ent, xh);
    k_transpose_w<<<dim3((HID * HID) / 256), dim3(256), 0, stream>>>(Wq, wtq);
    k_transpose_w<<<dim3((HID * HID) / 256), dim3(256), 0, stream>>>(Wk, wtk);
    k_transpose_w<<<dim3((HID * HID) / 256), dim3(256), 0, stream>>>(Wv, wtv);
    k_transpose_w<<<dim3((HID * HID) / 256), dim3(256), 0, stream>>>(Wo, wto);

    // 2) QKV projections (v stored transposed per head), 64x64 tiles per wave
    dim3 ggrid(MROWS / 64, NH);
    k_gemm_qkv<<<ggrid, dim3(32), 0, stream>>>(xh, wtq, bq, qh, 0);
    k_gemm_qkv<<<ggrid, dim3(32), 0, stream>>>(xh, wtk, bk, kh, 0);
    k_gemm_qkv<<<ggrid, dim3(32), 0, stream>>>(xh, wtv, bv, vTh, 1);

    // 3) attention
    k_attn<<<dim3(SEQ / 16, NH, BATCH), dim3(32), 0, stream>>>(qh, kh, vTh, mask, ctxh);

    // 4) output projection + residual, then LayerNorm + split
    k_outproj<<<ggrid, dim3(32), 0, stream>>>(ctxh, wto, bo, word, ent, yws);
    k_layernorm<<<dim3(MROWS), dim3(256), 0, stream>>>(yws, lnw, lnb, outp);
}
